// ReprogrammingEnc_84146999263424
// MI455X (gfx1250) — matrix operand order repaired and checked by reading, not compiled, not run on hardware
//
#include <hip/hip_runtime.h>
#include <hip/hip_bf16.h>
#include <stddef.h>

// ---------------------------------------------------------------------------
// MI455X (gfx1250) implementation.
//  B=4 L=2048 D_MODEL=1024 H=16 DK=64 D_LLM=4096 VOCAB=32000 S=1000
//  All heavy math runs through v_wmma_f32_16x16x32_bf16 (wave32 WMMA).
//  Pipeline:
//   1) source  = W_map @ word_emb + b_map          -> bf16 [1008,4096]
//   2) q       = enc  @ W_q^T + b_q                -> bf16 [8192,1024]
//   3) k,v     = source @ W_{k,v}^T + b            -> bf16 [1008,1024]
//   4) flash-attention per (b,h): K,V head in ~265KB LDS, online softmax
//   5) out     = attn @ W_o^T + b_o                -> f32 d_out [8192,4096]
//  LDS layouts are chosen so every WMMA fragment load is a pair of
//  contiguous 16B runs -> ds_load_b128 (no scalar u16 gathers).
// ---------------------------------------------------------------------------

typedef __attribute__((ext_vector_type(16))) __bf16        v16bf;
typedef __attribute__((ext_vector_type(8)))  float         v8f;
typedef __attribute__((ext_vector_type(8)))  unsigned short v8us;

union BfFrag { v16bf v; v8us h[2]; unsigned short u[16]; };
union Pack8  { v8us v; unsigned short u[8]; };

// fp32 -> bf16 (round to nearest even), bit-level
__device__ __forceinline__ unsigned short f2bf(float f) {
    unsigned u = __float_as_uint(f);
    u += 0x7FFFu + ((u >> 16) & 1u);
    return (unsigned short)(u >> 16);
}
__device__ __forceinline__ float bf2f(unsigned short h) {
    return __uint_as_float(((unsigned)h) << 16);
}

// --- A fragment, 16x32 bf16 (ISA 7.12.2): lane<16: row=lane, K {0..7,16..23};
//     lane>=16: row=lane-16, K {8..15,24..31}. src row-major, ld in elements.
//     Per-lane data = two contiguous 16B runs -> 2x ds_load_b128 / global_load_b128.
__device__ __forceinline__ v16bf load_a16x32(const unsigned short* __restrict__ src, int ld) {
    const int lane = threadIdx.x & 31;
    const int row  = lane & 15;
    const int kb   = (lane & 16) ? 8 : 0;
    const unsigned short* p = src + (size_t)row * ld + kb;
    BfFrag f;
    f.h[0] = *(const v8us*)p;          // K kb .. kb+7
    f.h[1] = *(const v8us*)(p + 16);   // K kb+16 .. kb+23
    return f.v;
}

// --- B fragment, 32x16 bf16, from K-contiguous ("transposed") storage:
//     element(k,n) = srcT[n*ldT + k]. lanes 0-15: col=lane, K {0..7,16..23};
//     lanes 16-31: K {8..15,24..31}. Per-lane = two contiguous 16B runs -> 2x b128.
__device__ __forceinline__ v16bf load_bT32x16(const unsigned short* __restrict__ srcT, int ldT) {
    const int lane = threadIdx.x & 31;
    const int n  = lane & 15;
    const int kb = (lane & 16) ? 8 : 0;
    const unsigned short* p = srcT + (size_t)n * ldT + kb;
    BfFrag f;
    f.h[0] = *(const v8us*)p;
    f.h[1] = *(const v8us*)(p + 16);
    return f.v;
}

// ---------------------------------------------------------------------------
// Generic tiled bf16 WMMA GEMM:  D[M,N] = A[M,K] * B[K,N] (+bias)
//  A: fp32 or bf16, row-major, rows >= a_rows read as 0
//  B: fp32; b_trans ? B(k,n)=Bp[n*ldb+k] : Bp[k*ldb+n]
//  Tile: 64(M) x 128(N) per 256-thread block (8 waves, each 16x64), K step 32.
//  B tile staged TRANSPOSED in LDS (Blt[n][k]) for contiguous fragment loads.
//  swap_xy: grid (m,n) instead of (n,m) -> co-schedules blocks sharing B
//  stripes (used for the 262-GFLOP embedding GEMM so W_map stays L2-resident
//  and the 524MB embedding table streams from HBM ~once).
// ---------------------------------------------------------------------------
#define TM 64
#define TN 128
#define TKC 32

__global__ __launch_bounds__(256) void gemm_bf16_wmma(
    const void* __restrict__ Ap, int a_bf16, int lda, int a_rows,
    const float* __restrict__ Bp, int b_trans, int ldb,
    void* __restrict__ Dp, int d_bf16, int ldd, int d_rows,
    const float* __restrict__ bias, int bias_mode, int bias_len,
    int N, int K, int swap_xy)
{
    __shared__ unsigned short Al[TM * TKC];    // 4 KB, row-major [m][k]
    __shared__ unsigned short Blt[TN * TKC];   // 8 KB, transposed [n][k]

    const int tid = threadIdx.x;
    const int m0  = (swap_xy ? blockIdx.x : blockIdx.y) * TM;
    const int n0  = (swap_xy ? blockIdx.y : blockIdx.x) * TN;
    const int wid = tid >> 5;
    const int wm  = wid & 3;        // 4 M-waves
    const int wn  = wid >> 2;       // 2 N-waves

    const float*          Af = (const float*)Ap;
    const unsigned short* Ab = (const unsigned short*)Ap;

    v8f acc[4] = {};

    for (int kc = 0; kc < K; kc += TKC) {   // K is a multiple of 32 in every launch
        // ---- stage A tile 64x32: each thread 8 contiguous K of one row
        {
            const int base = tid * 8;           // 2048 elements total
            const int r = base >> 5, c = base & 31;
            const int gr = m0 + r;
            Pack8 w;
            if (gr < a_rows) {
                const size_t off = (size_t)gr * lda + (kc + c);
                if (a_bf16) {
                    w.v = *(const v8us*)(Ab + off);
                } else {
#pragma unroll
                    for (int j = 0; j < 8; ++j) w.u[j] = f2bf(Af[off + j]);
                }
            } else {
#pragma unroll
                for (int j = 0; j < 8; ++j) w.u[j] = 0;
            }
            *(v8us*)(Al + base) = w.v;          // base = r*32 + c, 16B aligned
        }
        // ---- stage B tile 32x128 into transposed Blt[n][k]
        if (b_trans) {
            // global reads contiguous in k: thread handles 16 k for one n
            const int base = tid * 16;          // 4096 elements total
            const int n = base >> 5, k0 = base & 31;
            const int gn = n0 + n;
            Pack8 w0, w1;
            if (gn < N) {
                const float* bp = Bp + (size_t)gn * ldb + (kc + k0);
#pragma unroll
                for (int j = 0; j < 8; ++j) { w0.u[j] = f2bf(bp[j]); w1.u[j] = f2bf(bp[8 + j]); }
            } else {
#pragma unroll
                for (int j = 0; j < 8; ++j) { w0.u[j] = 0; w1.u[j] = 0; }
            }
            *(v8us*)(Blt + n * TKC + k0)     = w0.v;
            *(v8us*)(Blt + n * TKC + k0 + 8) = w1.v;
        } else {
            // global reads contiguous in n; scatter-transpose into LDS
            for (int idx = tid; idx < TKC * TN; idx += 256) {
                const int kk = idx >> 7, n = idx & 127;
                const int gn = n0 + n;
                const float val = (gn < N) ? Bp[(size_t)(kc + kk) * ldb + gn] : 0.f;
                Blt[n * TKC + kk] = f2bf(val);
            }
        }
        __syncthreads();

        const v16bf a = load_a16x32(Al + (wm * 16) * TKC, TKC);
#pragma unroll
        for (int t = 0; t < 4; ++t) {
            const v16bf b = load_bT32x16(Blt + (wn * 64 + t * 16) * TKC, TKC);
            acc[t] = __builtin_amdgcn_wmma_f32_16x16x32_bf16(
                false, a, false, b, (short)0, acc[t], false, false);
        }
        __syncthreads();
    }

    // store D (C/D layout: col = lane&15, VGPR r -> row r (lanes 0-15) / 8+r (16-31))
    const int lane  = tid & 31;
    const int cn    = lane & 15;
    const int rbase = (lane & 16) ? 8 : 0;
#pragma unroll
    for (int t = 0; t < 4; ++t) {
        const int gn = n0 + wn * 64 + t * 16 + cn;
        if (gn >= N) continue;
#pragma unroll
        for (int r = 0; r < 8; ++r) {
            const int gm = m0 + wm * 16 + rbase + r;
            if (gm >= d_rows) continue;
            float val = acc[t][r];
            if (bias_mode == 1)      val += bias[gn];
            else if (bias_mode == 2) val += bias[gm < bias_len ? gm : bias_len - 1];
            const size_t off = (size_t)gm * ldd + gn;
            if (d_bf16) ((unsigned short*)Dp)[off] = f2bf(val);
            else        ((float*)Dp)[off]          = val;
        }
    }
}

// ---------------------------------------------------------------------------
// Fused flash-attention over the prototype dimension (S=1000, padded 1024).
//  grid = (L/128, B*H), block = 256 (8 waves).
//  LDS: K as [s][64] (128KB, QK^T B-frags contiguous), V TRANSPOSED as
//  [e][VLD=1032] (~129KB, P.V B-frags contiguous, padded ld to spread banks),
//  8KB P-rotation scratch. Total ~265KB dynamic LDS (<320KB CDNA5 WGP).
//  Each wave: 16 q rows, online softmax over 32 chunks of 32 keys,
//  8 wmma_bf16 per chunk (4 QK^T + 4 P.V).
// ---------------------------------------------------------------------------
#define S_REAL 1000
#define S_PAD  1024
#define VLD    1032

__global__ __launch_bounds__(256) void attn_flash_wmma(
    const unsigned short* __restrict__ q,   // [8192,1024] bf16 (row=b*2048+l, col=h*64+e)
    const unsigned short* __restrict__ k,   // [1008,1024] bf16
    const unsigned short* __restrict__ v,   // [1008,1024] bf16
    unsigned short* __restrict__ o)         // [8192,1024] bf16
{
    extern __shared__ unsigned short sm[];
    unsigned short* Kl = sm;                           // [1024][64]
    unsigned short* Vt = sm + S_PAD * 64;              // [64][VLD] (transposed)
    unsigned short* Pl = sm + S_PAD * 64 + 64 * VLD;   // 8 waves * [16][32]

    const int tid = threadIdx.x;
    const int bh  = blockIdx.y;
    const int b   = bh >> 4;
    const int h   = bh & 15;

    // cooperative K/V head load (1008 valid rows; pad with 0). 8 elems/thread/iter.
    for (int idx = tid * 8; idx < S_PAD * 64; idx += 256 * 8) {
        const int s = idx >> 6, e = idx & 63;
        Pack8 wk, wv;
        if (s < 1008) {
            const size_t off = (size_t)s * 1024 + h * 64 + e;
            wk.v = *(const v8us*)(k + off);
            wv.v = *(const v8us*)(v + off);
        } else {
#pragma unroll
            for (int j = 0; j < 8; ++j) { wk.u[j] = 0; wv.u[j] = 0; }
        }
        *(v8us*)(Kl + idx) = wk.v;          // [s][e] row-major
#pragma unroll
        for (int j = 0; j < 8; ++j)          // transpose-scatter V -> [e][s]
            Vt[(e + j) * VLD + s] = wv.u[j];
    }
    __syncthreads();

    const int wid  = tid >> 5;
    const int lane = tid & 31;
    const int l0   = blockIdx.x * 128 + wid * 16;
    const size_t qb = ((size_t)(b * 2048 + l0)) * 1024 + h * 64;

    const v16bf a0 = load_a16x32(q + qb, 1024);        // e 0..31
    const v16bf a1 = load_a16x32(q + qb + 32, 1024);   // e 32..63

    float mrow[8], lrow[8];
#pragma unroll
    for (int r = 0; r < 8; ++r) { mrow[r] = -1e30f; lrow[r] = 0.f; }
    v8f oacc[4] = {};
    const float scale = 0.125f;                         // 1/sqrt(64)
    unsigned short* Pw = Pl + wid * 16 * 32;
    const int cn    = lane & 15;
    const int rbase = (lane & 16) ? 8 : 0;

    for (int sc = 0; sc < S_PAD / 32; ++sc) {
        const int sbase = sc * 32;
        float p[2][8];
        float cmax[8];
#pragma unroll
        for (int r = 0; r < 8; ++r) cmax[r] = -1e30f;

        // scores for 32 keys: two 16-col WMMA C tiles, each accumulating e=0..63.
        // B(k=e,n=s) = Kl[(s0+n)*64 + e] -> contiguous per lane (ldT=64).
#pragma unroll
        for (int sub = 0; sub < 2; ++sub) {
            v8f c = {};
            const v16bf bk0 = load_bT32x16(Kl + (sbase + sub * 16) * 64,      64);
            const v16bf bk1 = load_bT32x16(Kl + (sbase + sub * 16) * 64 + 32, 64);
            c = __builtin_amdgcn_wmma_f32_16x16x32_bf16(false, a0, false, bk0, (short)0, c, false, false);
            c = __builtin_amdgcn_wmma_f32_16x16x32_bf16(false, a1, false, bk1, (short)0, c, false, false);
            const int sg = sbase + sub * 16 + cn;
#pragma unroll
            for (int r = 0; r < 8; ++r) {
                const float val = (sg < S_REAL) ? c[r] * scale : -1e30f;
                p[sub][r] = val;
                cmax[r]   = fmaxf(cmax[r], val);
            }
        }
        // row max across the 16 lanes of each half (rows 0-7 | 8-15 stay separate)
#pragma unroll
        for (int off = 1; off <= 8; off <<= 1)
#pragma unroll
            for (int r = 0; r < 8; ++r)
                cmax[r] = fmaxf(cmax[r], __shfl_xor(cmax[r], off, 32));

        float alpha[8], psum[8];
#pragma unroll
        for (int r = 0; r < 8; ++r) {
            const float mn = fmaxf(mrow[r], cmax[r]);
            alpha[r] = __expf(mrow[r] - mn);
            mrow[r]  = mn;
            psum[r]  = 0.f;
        }
#pragma unroll
        for (int sub = 0; sub < 2; ++sub)
#pragma unroll
            for (int r = 0; r < 8; ++r) {
                const float pe = (p[sub][r] > -1e29f) ? __expf(p[sub][r] - mrow[r]) : 0.f;
                p[sub][r] = pe;
                psum[r]  += pe;
            }
#pragma unroll
        for (int off = 1; off <= 8; off <<= 1)
#pragma unroll
            for (int r = 0; r < 8; ++r)
                psum[r] += __shfl_xor(psum[r], off, 32);
#pragma unroll
        for (int r = 0; r < 8; ++r) lrow[r] = lrow[r] * alpha[r] + psum[r];
#pragma unroll
        for (int t = 0; t < 4; ++t)
#pragma unroll
            for (int r = 0; r < 8; ++r) oacc[t][r] *= alpha[r];

        // rotate P from C-layout to A-layout via per-wave LDS (in-order within wave)
#pragma unroll
        for (int sub = 0; sub < 2; ++sub)
#pragma unroll
            for (int r = 0; r < 8; ++r)
                Pw[(rbase + r) * 32 + sub * 16 + cn] = f2bf(p[sub][r]);

        const v16bf ap = load_a16x32(Pw, 32);
        // B(k=s_local, n=e) = Vt[(t*16+n)*VLD + sbase + k] -> contiguous per lane.
#pragma unroll
        for (int t = 0; t < 4; ++t) {
            const v16bf bv = load_bT32x16(Vt + (t * 16) * VLD + sbase, VLD);
            oacc[t] = __builtin_amdgcn_wmma_f32_16x16x32_bf16(
                false, ap, false, bv, (short)0, oacc[t], false, false);
        }
    }

    // normalize and store
    const size_t ob = ((size_t)(b * 2048 + l0)) * 1024 + h * 64;
#pragma unroll
    for (int r = 0; r < 8; ++r) {
        const float inv = (lrow[r] > 0.f) ? 1.f / lrow[r] : 0.f;
#pragma unroll
        for (int t = 0; t < 4; ++t)
            o[ob + (size_t)(rbase + r) * 1024 + t * 16 + cn] = f2bf(oacc[t][r] * inv);
    }
}

// ---------------------------------------------------------------------------
// Host launcher
// ---------------------------------------------------------------------------
extern "C" void kernel_launch(void* const* d_in, const int* in_sizes, int n_in,
                              void* d_out, int out_size, void* d_ws, size_t ws_size,
                              hipStream_t stream) {
    (void)in_sizes; (void)n_in; (void)out_size; (void)ws_size;

    const float* enc  = (const float*)d_in[0];   // [4,2048,1024]
    const float* wemb = (const float*)d_in[1];   // [32000,4096]
    const float* Wmap = (const float*)d_in[2];   // [1000,32000]
    const float* bmap = (const float*)d_in[3];   // [1000]
    const float* Wq   = (const float*)d_in[4];   // [1024,1024]
    const float* bq   = (const float*)d_in[5];
    const float* Wk   = (const float*)d_in[6];   // [1024,4096]
    const float* bk   = (const float*)d_in[7];
    const float* Wv   = (const float*)d_in[8];   // [1024,4096]
    const float* bv   = (const float*)d_in[9];
    const float* Wo   = (const float*)d_in[10];  // [4096,1024]
    const float* bo   = (const float*)d_in[11];

    // workspace layout (bf16 = ushort). Needs ~46 MB.
    unsigned short* src_bf = (unsigned short*)d_ws;        // [1008,4096]
    unsigned short* q_bf   = src_bf + (size_t)1008 * 4096; // [8192,1024]
    unsigned short* k_bf   = q_bf   + (size_t)8192 * 1024; // [1008,1024]
    unsigned short* v_bf   = k_bf   + (size_t)1008 * 1024; // [1008,1024]
    unsigned short* at_bf  = v_bf   + (size_t)1008 * 1024; // [8192,1024]

    const dim3 blk(256);

    // 1) source = W_map @ word_emb + b_map (per-row bias) -> bf16 [1008,4096]
    //    swap_xy=1: M-blocks innermost so all 16 blocks sharing a wemb stripe
    //    run together; W_map (128MB fp32) stays L2-resident, wemb streams once.
    gemm_bf16_wmma<<<dim3(1024 / TM, 4096 / TN), blk, 0, stream>>>(
        Wmap, 0, 32000, 1000,
        wemb, 0, 4096,
        src_bf, 1, 4096, 1008,
        bmap, 2, 1000,
        4096, 32000, 1);

    // 2) q = enc @ W_q^T + b_q -> bf16 [8192,1024]
    gemm_bf16_wmma<<<dim3(1024 / TN, 8192 / TM), blk, 0, stream>>>(
        enc, 0, 1024, 8192,
        Wq, 1, 1024,
        q_bf, 1, 1024, 8192,
        bq, 1, 1024,
        1024, 1024, 0);

    // 3) k = source @ W_k^T + b_k ; v = source @ W_v^T + b_v -> bf16 [1008,1024]
    gemm_bf16_wmma<<<dim3(1024 / TN, 1024 / TM), blk, 0, stream>>>(
        src_bf, 1, 4096, 1008,
        Wk, 1, 4096,
        k_bf, 1, 1024, 1008,
        bk, 1, 1024,
        1024, 4096, 0);
    gemm_bf16_wmma<<<dim3(1024 / TN, 1024 / TM), blk, 0, stream>>>(
        src_bf, 1, 4096, 1008,
        Wv, 1, 4096,
        v_bf, 1, 1024, 1008,
        bv, 1, 1024,
        1024, 4096, 0);

    // 4) fused attention: grid (L/128, B*H); ~265 KB dynamic LDS (CDNA5: 320 KB/WGP)
    const size_t lds_bytes = (size_t)(S_PAD * 64 + 64 * VLD + 8 * 16 * 32) * sizeof(unsigned short);
    attn_flash_wmma<<<dim3(2048 / 128, 4 * 16), blk, lds_bytes, stream>>>(
        q_bf, k_bf, v_bf, at_bf);

    // 5) out = attn @ W_o^T + b_o -> f32 [8192,4096]
    gemm_bf16_wmma<<<dim3(4096 / TN, 8192 / TM), blk, 0, stream>>>(
        at_bf, 1, 1024, 8192,
        Wo, 1, 1024,
        d_out, 0, 4096, 8192,
        bo, 1, 4096,
        4096, 1024, 0);
}
